// PriorNetwork_29798483100030
// MI455X (gfx1250) — matrix operand order repaired and checked by reading, not compiled, not run on hardware
//
#include <hip/hip_runtime.h>
#include <hip/hip_bf16.h>

// ---------------------------------------------------------------------------
// PriorNetwork forward for MI455X (gfx1250, wave32, WMMA).
// Pipeline:
//   K0  train_prep : train f32 -> f16, row sum-of-squares t2
//   K1  cvt_f16    : codes / fc1_w / fc2u_w / fc2s_w -> f16
//   K2  knn_chunk  : f16 WMMA scores (t2 - 2*q.t); B-tiles staged to LDS with
//                    global_load_async_to_lds_b128 (double-buffered, ASYNCcnt),
//                    shared by 4 waves (64 FLOP/B from L2); top-8/chunk cands
//   K3  rerank     : exact fp32 distances on candidates, sorted top-5, select
//   K4  mlp        : fused 2-layer MLP (f16 WMMA, f32 accum), writes mu|logstd
// ---------------------------------------------------------------------------

typedef _Float16 v16h __attribute__((ext_vector_type(16)));
typedef float    v8f  __attribute__((ext_vector_type(8)));
typedef int      i4   __attribute__((ext_vector_type(4)));

union F16x16 { v16h v; i4 i[2]; };

#define FMAXV 3.402823466e38f
#define DWIDTH 64          // code dim D
#define HWIDTH 512         // hidden H
#define CHUNK_COLS 512     // train columns per block in knn stage
#define PH_ROWS 64         // train rows staged per phase (8 KB)
#define NPH (CHUNK_COLS / PH_ROWS)   // 8 phases
#define TOPC 8             // candidates kept per (query, chunk)

static __device__ __forceinline__ v8f wmma_f16(const F16x16& a, const F16x16& b, v8f c) {
  return __builtin_amdgcn_wmma_f32_16x16x32_f16(false, a.v, false, b.v, (short)0, c, false, false);
}

// 4 x b128 async global->LDS copies; imm offset advances both LDS and global
// addresses (ISA 08_async_tensor 4.4). One call = exactly 4 ASYNCcnt ticks.
static __device__ __forceinline__ void async_b128x4(unsigned lds_addr, unsigned goff,
                                                    unsigned long long base) {
  asm volatile(
      "global_load_async_to_lds_b128 %0, %1, %2 offset:0\n\t"
      "global_load_async_to_lds_b128 %0, %1, %2 offset:16\n\t"
      "global_load_async_to_lds_b128 %0, %1, %2 offset:32\n\t"
      "global_load_async_to_lds_b128 %0, %1, %2 offset:48"
      :: "v"(lds_addr), "v"(goff), "s"(base) : "memory");
}

// Stage PH_ROWS train rows (128 B each) into LDS: thread = (row, half-row seg).
// OOB rows are clamped to row 0 (kept uniform so ASYNCcnt stays exact) and are
// masked later via t2j = FLT_MAX.
static __device__ __forceinline__ void issue_phase(_Float16* buf, int colp, int pr,
                                                   int seg, int N,
                                                   unsigned long long t16base) {
  int j = colp + pr;
  if (j >= N) j = 0;
  unsigned goff = (unsigned)j * 128u + (unsigned)seg * 64u;
  unsigned ldsa = (unsigned)(size_t)buf + (unsigned)(pr * 128 + seg * 64);
  async_b128x4(ldsa, goff, t16base);
}

// ---------------------------------------------------------------- K0 / K1 ---
__global__ void train_prep_kernel(const float* __restrict__ tc,
                                  _Float16* __restrict__ t16,
                                  float* __restrict__ t2, int N) {
  int r = blockIdx.x * blockDim.x + threadIdx.x;
  if (r >= N) return;
  const float* row = tc + (size_t)r * DWIDTH;
  _Float16* dst = t16 + (size_t)r * DWIDTH;
  float acc = 0.f;
#pragma unroll
  for (int k = 0; k < DWIDTH; ++k) {
    float v = row[k];
    acc = fmaf(v, v, acc);
    dst[k] = (_Float16)v;
  }
  t2[r] = acc;
}

__global__ void cvt_f16_kernel(const float* __restrict__ src,
                               _Float16* __restrict__ dst, int n) {
  int i = blockIdx.x * blockDim.x + threadIdx.x;
  if (i < n) dst[i] = (_Float16)src[i];
}

// --------------------------------------------------------------------- K2 ---
// grid = (nchunk, B/64), block = 128 (4 waves). Each wave owns one 16-query
// A-tile; all 4 waves consume the same async-staged B-tiles (512 columns per
// block, 64 rows per double-buffered phase). Per-(lane,row) top-5 lists in LDS
// behind a register threshold with one min-chain branch per tile.
__global__ void knn_chunk_kernel(const _Float16* __restrict__ q16,
                                 const _Float16* __restrict__ t16,
                                 const float* __restrict__ t2,
                                 int* __restrict__ cand,
                                 int N, int nchunk) {
  __shared__ _Float16 bstage[2][PH_ROWS * DWIDTH];   // 2 x 8 KB
  __shared__ float ls[128 * 8 * 5];                  // 20 KB
  __shared__ int   li[128 * 8 * 5];                  // 20 KB

  const int tid = threadIdx.x;
  const int lane = tid & 31, wv = tid >> 5;
  const int half = lane >> 4, ln = lane & 15;
  const int chunk = blockIdx.x, qblk = blockIdx.y;

  {
    int base = tid * 40;
    for (int e = 0; e < 40; ++e) { ls[base + e] = FMAXV; li[base + e] = -1; }
  }
  float thr[8];
#pragma unroll
  for (int r = 0; r < 8; ++r) thr[r] = FMAXV;

  // A fragments: this wave's 16 query rows, K = 0..63.
  F16x16 a0, a1;
  {
    const _Float16* qrow = q16 + (size_t)(qblk * 64 + wv * 16 + ln) * DWIDTH;
    a0.i[0] = *(const i4*)(qrow + 8 * half);
    a0.i[1] = *(const i4*)(qrow + 8 * half + 16);
    a1.i[0] = *(const i4*)(qrow + 32 + 8 * half);
    a1.i[1] = *(const i4*)(qrow + 32 + 8 * half + 16);
  }

  const int pr = tid >> 1, seg = tid & 1;
  const unsigned long long tb = (unsigned long long)(size_t)t16;
  const int colchunk = chunk * CHUNK_COLS;

  issue_phase(&bstage[0][0], colchunk, pr, seg, N, tb);
  for (int p = 0; p < NPH; ++p) {
    if (p + 1 < NPH) {
      issue_phase(&bstage[(p + 1) & 1][0], colchunk + (p + 1) * PH_ROWS, pr, seg, N, tb);
      asm volatile("s_wait_asynccnt 0x4" ::: "memory");  // phase p landed
    } else {
      asm volatile("s_wait_asynccnt 0x0" ::: "memory");
    }
    __syncthreads();
    const _Float16* buf = &bstage[p & 1][0];
#pragma unroll
    for (int tt = 0; tt < 4; ++tt) {
      const int colbase = colchunk + p * PH_ROWS + tt * 16;
      const int j = colbase + ln;
      const _Float16* brow = buf + (tt * 16 + ln) * DWIDTH;
      F16x16 b0, b1;
      b0.i[0] = *(const i4*)(brow + 8 * half);
      b0.i[1] = *(const i4*)(brow + 8 * half + 16);
      b1.i[0] = *(const i4*)(brow + 32 + 8 * half);
      b1.i[1] = *(const i4*)(brow + 32 + 8 * half + 16);
      float t2j = (j < N) ? t2[j] : FMAXV;  // OOB columns can never insert
      v8f c = {};
      c = wmma_f16(a0, b0, c);
      c = wmma_f16(a1, b1, c);
      float s[8], worst;
#pragma unroll
      for (int r = 0; r < 8; ++r) {
        s[r] = fmaf(-2.0f, c[r], t2j);      // rank-equivalent to d2
        float d = s[r] - thr[r];
        worst = (r == 0) ? d : fminf(worst, d);
      }
      if (worst < 0.f) {
#pragma unroll
        for (int r = 0; r < 8; ++r) {
          if (s[r] < thr[r]) {
            int base = (tid * 8 + r) * 5;
            int mi = 0; float mx = ls[base];
#pragma unroll
            for (int e = 1; e < 5; ++e) { float v = ls[base + e]; if (v > mx) { mx = v; mi = e; } }
            ls[base + mi] = s[r]; li[base + mi] = j;
            float nm = ls[base];
#pragma unroll
            for (int e = 1; e < 5; ++e) { float v = ls[base + e]; if (v > nm) nm = v; }
            thr[r] = nm;
          }
        }
      }
    }
    __syncthreads();
  }

  // Merge: each query is owned by one wave -> 16 lanes x 5 entries -> top-8.
  if (tid < 64) {
    const int m = tid;                         // query within block (0..63)
    const int wq = m >> 4, mm = m & 15;
    const int r = mm & 7, hh = mm >> 3;
    int* outp = cand + ((size_t)(qblk * 64 + m) * nchunk + chunk) * TOPC;
    for (int rank = 0; rank < TOPC; ++rank) {
      float best = FMAXV; int bi = -1, bslot = -1;
      for (int l2 = 0; l2 < 16; ++l2) {
        int t2id = wq * 32 + hh * 16 + l2;
        int base = (t2id * 8 + r) * 5;
#pragma unroll
        for (int e = 0; e < 5; ++e) {
          float v = ls[base + e]; int ix = li[base + e];
          if (v < best || (v == best && (unsigned)ix < (unsigned)bi)) {
            best = v; bi = ix; bslot = base + e;
          }
        }
      }
      outp[rank] = bi;
      if (bslot >= 0) { ls[bslot] = FMAXV; li[bslot] = -1; }
    }
  }
}

// --------------------------------------------------------------------- K3 ---
// grid = B blocks, block = 128. Exact fp32 rerank of candidates, sorted top-5.
__global__ void rerank_select_kernel(const float* __restrict__ codes,
                                     const float* __restrict__ train,
                                     const int* __restrict__ cand,
                                     const int* __restrict__ choice,
                                     int* __restrict__ sel,
                                     int N, int nchunk) {
  __shared__ float q[DWIDTH];
  __shared__ float dsv[2048];
  __shared__ int   div_[2048];
  const int b = blockIdx.x, tid = threadIdx.x;
  const int NC = nchunk * TOPC;   // <= 2048
  if (tid < DWIDTH) q[tid] = codes[(size_t)b * DWIDTH + tid];
  __syncthreads();
  for (int i = tid; i < NC; i += blockDim.x) {
    int ix = cand[(size_t)b * NC + i];
    float d = FMAXV;
    if (ix >= 0) {
      const float* tr = train + (size_t)ix * DWIDTH;
      float acc = 0.f;
#pragma unroll
      for (int k = 0; k < DWIDTH; ++k) { float df = q[k] - tr[k]; acc = fmaf(df, df, acc); }
      d = acc;
    }
    dsv[i] = d; div_[i] = ix;
  }
  __syncthreads();
  if (tid == 0) {
    int top[5];
    for (int rank = 0; rank < 5; ++rank) {
      float best = FMAXV; int bi = -1, bs = -1;
      for (int i = 0; i < NC; ++i) {
        float v = dsv[i]; int ix = div_[i];
        if (v < best || (v == best && (unsigned)ix < (unsigned)bi)) { best = v; bi = ix; bs = i; }
      }
      top[rank] = bi;
      if (bs >= 0) { dsv[bs] = FMAXV; div_[bs] = -1; }
    }
    int c = choice[b]; c = c < 0 ? 0 : (c > 4 ? 4 : c);
    sel[b] = top[c] < 0 ? 0 : top[c];
  }
}

// --------------------------------------------------------------------- K4 ---
// grid = B/16 blocks, block = 128 (4 waves). Fused MLP: h1 tile in LDS,
// fc1 split over N across waves, heads split over output tiles across waves.
__global__ void mlp_kernel(const _Float16* __restrict__ t16,
                           const int* __restrict__ sel,
                           const _Float16* __restrict__ w1h,
                           const float* __restrict__ fb1,
                           const _Float16* __restrict__ w2uh,
                           const float* __restrict__ fb2u,
                           const _Float16* __restrict__ w2sh,
                           const float* __restrict__ fb2s,
                           float* __restrict__ out, int B) {
  __shared__ _Float16 prev[16 * DWIDTH];
  __shared__ _Float16 h16[16 * HWIDTH];
  const int tid = threadIdx.x, lane = tid & 31, wv = tid >> 5;
  const int half = lane >> 4, ln = lane & 15;
  const int qtile = blockIdx.x;

  for (int i = tid; i < 16 * DWIDTH; i += 128) {
    int m = i >> 6, k = i & 63;
    int s = sel[qtile * 16 + m];
    prev[i] = t16[(size_t)s * DWIDTH + k];
  }
  __syncthreads();

  // A fragments of prev_code tile (LDS -> ds_load_b128).
  F16x16 a0, a1;
  {
    const _Float16* row = prev + ln * DWIDTH;
    a0.i[0] = *(const i4*)(row + 8 * half);
    a0.i[1] = *(const i4*)(row + 8 * half + 16);
    a1.i[0] = *(const i4*)(row + 32 + 8 * half);
    a1.i[1] = *(const i4*)(row + 32 + 8 * half + 16);
  }

  // fc1: wave w handles hidden cols [w*128, w*128+128)
  for (int ct = 0; ct < 8; ++ct) {
    int n = wv * 128 + ct * 16 + ln;
    const _Float16* wrow = w1h + (size_t)n * DWIDTH;
    F16x16 b0, b1;
    b0.i[0] = *(const i4*)(wrow + 8 * half);
    b0.i[1] = *(const i4*)(wrow + 8 * half + 16);
    b1.i[0] = *(const i4*)(wrow + 32 + 8 * half);
    b1.i[1] = *(const i4*)(wrow + 32 + 8 * half + 16);
    v8f c = {};
    c = wmma_f16(a0, b0, c);
    c = wmma_f16(a1, b1, c);
    float bias = fb1[n];
#pragma unroll
    for (int r = 0; r < 8; ++r) {
      float h = c[r] + bias;
      h = h > 0.f ? h : 0.f;
      h16[(r + 8 * half) * HWIDTH + n] = (_Float16)h;
    }
  }
  __syncthreads();

  // Heads: 8 output col-tiles (mu: 0..3, logstd: 4..7), 2 per wave. K = 512.
  for (int tt = 0; tt < 2; ++tt) {
    int T = wv * 2 + tt;
    int head = T >> 2, nt = T & 3;
    const _Float16* W = head ? w2sh : w2uh;
    const float* Bv = head ? fb2s : fb2u;
    int n = nt * 16 + ln;
    const _Float16* wrow = W + (size_t)n * HWIDTH;
    v8f c = {};
    for (int kk = 0; kk < 16; ++kk) {
      F16x16 af, bf;
      const _Float16* hrow = h16 + ln * HWIDTH + kk * 32;
      af.i[0] = *(const i4*)(hrow + 8 * half);
      af.i[1] = *(const i4*)(hrow + 8 * half + 16);
      bf.i[0] = *(const i4*)(wrow + kk * 32 + 8 * half);
      bf.i[1] = *(const i4*)(wrow + kk * 32 + 8 * half + 16);
      c = wmma_f16(af, bf, c);
    }
    float bias = Bv[n];
    float* outp = out + (size_t)head * B * DWIDTH;
#pragma unroll
    for (int r = 0; r < 8; ++r) {
      int gm = qtile * 16 + r + 8 * half;
      outp[(size_t)gm * DWIDTH + n] = c[r] + bias;
    }
  }
}

// ------------------------------------------------------------------- host ---
extern "C" void kernel_launch(void* const* d_in, const int* in_sizes, int n_in,
                              void* d_out, int out_size, void* d_ws, size_t ws_size,
                              hipStream_t stream) {
  const float* codes  = (const float*)d_in[0];
  const float* train  = (const float*)d_in[1];
  const float* fc1w   = (const float*)d_in[2];
  const float* fc1b   = (const float*)d_in[3];
  const float* fc2uw  = (const float*)d_in[4];
  const float* fc2ub  = (const float*)d_in[5];
  const float* fc2sw  = (const float*)d_in[6];
  const float* fc2sb  = (const float*)d_in[7];
  const int*   choice = (const int*)d_in[8];

  const int B = in_sizes[0] / DWIDTH;   // 1024
  const int N = in_sizes[1] / DWIDTH;   // 100000
  const int H = in_sizes[3];            // 512
  const int nchunk = (N + CHUNK_COLS - 1) / CHUNK_COLS;

  char* ws = (char*)d_ws;
  size_t off = 0;
  auto alloc = [&](size_t bytes) -> void* {
    void* p = ws + off;
    off = (off + bytes + 255) & ~(size_t)255;
    return p;
  };
  _Float16* t16  = (_Float16*)alloc((size_t)N * DWIDTH * 2);
  _Float16* q16  = (_Float16*)alloc((size_t)B * DWIDTH * 2);
  float*    t2   = (float*)   alloc((size_t)N * 4);
  _Float16* w1h  = (_Float16*)alloc((size_t)H * DWIDTH * 2);
  _Float16* w2uh = (_Float16*)alloc((size_t)DWIDTH * H * 2);
  _Float16* w2sh = (_Float16*)alloc((size_t)DWIDTH * H * 2);
  int*      cand = (int*)     alloc((size_t)B * nchunk * TOPC * 4);
  int*      sel  = (int*)     alloc((size_t)B * 4);
  (void)ws_size; (void)n_in; (void)out_size;

  hipLaunchKernelGGL(train_prep_kernel, dim3((N + 127) / 128), dim3(128), 0, stream,
                     train, t16, t2, N);
  hipLaunchKernelGGL(cvt_f16_kernel, dim3((B * DWIDTH + 255) / 256), dim3(256), 0, stream,
                     codes, q16, B * DWIDTH);
  hipLaunchKernelGGL(cvt_f16_kernel, dim3((H * DWIDTH + 255) / 256), dim3(256), 0, stream,
                     fc1w, w1h, H * DWIDTH);
  hipLaunchKernelGGL(cvt_f16_kernel, dim3((DWIDTH * H + 255) / 256), dim3(256), 0, stream,
                     fc2uw, w2uh, DWIDTH * H);
  hipLaunchKernelGGL(cvt_f16_kernel, dim3((DWIDTH * H + 255) / 256), dim3(256), 0, stream,
                     fc2sw, w2sh, DWIDTH * H);
  hipLaunchKernelGGL(knn_chunk_kernel, dim3(nchunk, B / 64), dim3(128), 0, stream,
                     q16, t16, t2, cand, N, nchunk);
  hipLaunchKernelGGL(rerank_select_kernel, dim3(B), dim3(128), 0, stream,
                     codes, train, cand, choice, sel, N, nchunk);
  hipLaunchKernelGGL(mlp_kernel, dim3(B / 16), dim3(128), 0, stream,
                     t16, sel, w1h, fc1b, w2uh, fc2ub, w2sh, fc2sb, (float*)d_out, B);
}
